// CGCNN_58600533786659
// MI455X (gfx1250) — compile-verified, matrix operand order checked by reading, not run on hardware
//
#include <hip/hip_runtime.h>
#include <hip/hip_bf16.h>
#include <math.h>

#define N_NODES    50000
#define N_EDGES    1600000
#define ATOM_F     64
#define EDGE_F     128
#define DEPTH      3
#define NUM_GRAPHS 256
#define HIDDEN     128
#define KDIM       256           // 2*ATOM_F + EDGE_F
#define NDIM       128           // 2*ATOM_F
#define TILE_M     256           // edges per workgroup (8 waves x 32 edges)
#define EPS        1e-8f

typedef __attribute__((ext_vector_type(16))) __bf16 v16bf;
typedef __attribute__((ext_vector_type(8)))  __bf16 bf16x8;
typedef __attribute__((ext_vector_type(8)))  float  v8f;
typedef __attribute__((ext_vector_type(4)))  int    v4i;

typedef __attribute__((address_space(1))) v4i gv4i;   // global
typedef __attribute__((address_space(3))) v4i lv4i;   // LDS

union V16U { v16bf v; bf16x8 h[2]; };
union PackU { __bf16 h[8]; uint4 u; };

#if defined(__has_builtin)
#if __has_builtin(__builtin_amdgcn_global_load_async_to_lds_b128) && \
    __has_builtin(__builtin_amdgcn_s_wait_asynccnt)
#define USE_ASYNC_LDS 1
#endif
#endif
#ifndef USE_ASYNC_LDS
#define USE_ASYNC_LDS 0
#endif

// ---------------------------------------------------------------- utilities
__global__ void zero_f32(float* p, int n) {
    int i = blockIdx.x * blockDim.x + threadIdx.x;
    if (i < n) p[i] = 0.0f;
}

// x[n][f] = embed_table[node_fea[n]][f]
__global__ void embed_kernel(const int* __restrict__ node_fea,
                             const float* __restrict__ table,
                             float* __restrict__ x) {
    int i = blockIdx.x * blockDim.x + threadIdx.x;
    if (i >= N_NODES * ATOM_F) return;
    int n = i >> 6, f = i & 63;
    x[i] = table[node_fea[n] * ATOM_F + f];
}

// fc_w (DEPTH*128*256 fp32) -> bf16
__global__ void cvt_w_kernel(const float* __restrict__ w, __bf16* __restrict__ wb, int n) {
    int i = blockIdx.x * blockDim.x + threadIdx.x;
    if (i < n) wb[i] = (__bf16)w[i];
}

__device__ __forceinline__ void pack8(const float4 a, const float4 b, __bf16* dst) {
    PackU pk;
    pk.h[0] = (__bf16)a.x; pk.h[1] = (__bf16)a.y;
    pk.h[2] = (__bf16)a.z; pk.h[3] = (__bf16)a.w;
    pk.h[4] = (__bf16)b.x; pk.h[5] = (__bf16)b.y;
    pk.h[6] = (__bf16)b.z; pk.h[7] = (__bf16)b.w;
    *(uint4*)dst = pk.u;
}

// ------------------------------------------------ edge GEMM + gate + scatter
// block: 256 threads (8 waves). 256 edges per block, each wave 2 M-subtiles.
__global__ __launch_bounds__(256, 1)
void edge_conv_kernel(const float* __restrict__ x,
                      const int* __restrict__ src,
                      const int* __restrict__ dst,
                      const float* __restrict__ edge_fea,
                      const __bf16* __restrict__ w_bf,   // [128][256] layer weights
                      const float* __restrict__ fc_b,    // [128] layer bias
                      float* __restrict__ agg) {
    __shared__ __align__(16) __bf16 sA[TILE_M * KDIM];   // 128 KB
    __shared__ __align__(16) __bf16 sB[NDIM * KDIM];     //  64 KB

    const int tid = threadIdx.x;
    const int e0  = blockIdx.x * TILE_M;

    // ---- stage B: 64 KB bf16 weight tile
#if USE_ASYNC_LDS
    {
        // 65536 B / (256 lanes * 16 B) = 16 async b128 issues per thread
        char* g = (char*)w_bf + tid * 16;
        char* l = (char*)sB + tid * 16;
        #pragma unroll
        for (int i = 0; i < 16; ++i)
            __builtin_amdgcn_global_load_async_to_lds_b128(
                (gv4i*)(g + i * 4096), (lv4i*)(l + i * 4096), 0, 0);
    }
#else
    {
        const uint4* wsrc = (const uint4*)w_bf;
        uint4* wdst = (uint4*)sB;
        // 128*256 bf16 = 4096 uint4; loop copies 256 threads x 32 = 8192
        #pragma unroll
        for (int j = 0; j < 32; ++j)
            wdst[tid + j * 256] = wsrc[tid + j * 256];
    }
#endif

    // ---- stage A: one thread per edge: [x[src] | x[dst] | edge_fea] -> bf16
    {
        const int e = e0 + tid;
        __bf16* row = &sA[tid * KDIM];
        const float4* ps = (const float4*)&x[src[e] * ATOM_F];
        const float4* pd = (const float4*)&x[dst[e] * ATOM_F];
        const float4* pe = (const float4*)&edge_fea[(size_t)e * EDGE_F];
        #pragma unroll
        for (int j = 0; j < 8; ++j) pack8(ps[2 * j], ps[2 * j + 1], &row[8 * j]);
        #pragma unroll
        for (int j = 0; j < 8; ++j) pack8(pd[2 * j], pd[2 * j + 1], &row[ATOM_F + 8 * j]);
        #pragma unroll
        for (int j = 0; j < 16; ++j) pack8(pe[2 * j], pe[2 * j + 1], &row[2 * ATOM_F + 8 * j]);
    }
#if USE_ASYNC_LDS
    __builtin_amdgcn_s_wait_asynccnt(0);
#endif
    __syncthreads();

    // ---- WMMA: each wave -> 32 edges (2 subtiles of 16) x 128 outputs
    const int lane  = tid & 31;
    const int wid   = tid >> 5;
    const int row16 = lane & 15;
    const int khalf = (lane >> 4) << 3;       // 0 or 8 (ISA A/B half-lane K layout)

    v8f acc0[8], acc1[8];
    #pragma unroll
    for (int nt = 0; nt < 8; ++nt) {
        acc0[nt] = (v8f){0.f,0.f,0.f,0.f,0.f,0.f,0.f,0.f};
        acc1[nt] = (v8f){0.f,0.f,0.f,0.f,0.f,0.f,0.f,0.f};
    }

    const __bf16* aBase0 = &sA[(wid * 32 + row16) * KDIM + khalf];
    const __bf16* aBase1 = aBase0 + 16 * KDIM;
    #pragma unroll
    for (int kt = 0; kt < 8; ++kt) {
        V16U a0, a1;
        a0.h[0] = *(const bf16x8*)&aBase0[kt * 32];
        a0.h[1] = *(const bf16x8*)&aBase0[kt * 32 + 16];
        a1.h[0] = *(const bf16x8*)&aBase1[kt * 32];
        a1.h[1] = *(const bf16x8*)&aBase1[kt * 32 + 16];
        #pragma unroll
        for (int nt = 0; nt < 8; ++nt) {
            V16U b;
            const __bf16* bBase = &sB[(nt * 16 + row16) * KDIM + khalf];
            b.h[0] = *(const bf16x8*)&bBase[kt * 32];
            b.h[1] = *(const bf16x8*)&bBase[kt * 32 + 16];
            acc0[nt] = __builtin_amdgcn_wmma_f32_16x16x32_bf16(
                false, a0.v, false, b.v, (short)0, acc0[nt], false, false);
            acc1[nt] = __builtin_amdgcn_wmma_f32_16x16x32_bf16(
                false, a1.v, false, b.v, (short)0, acc1[nt], false, false);
        }
    }

    // ---- gate/core, activation, scatter-add into agg[dst]
    // C frag: lane -> N = nt*16 + row16 ; VGPR r -> M = r + (lane>=16 ? 8 : 0)
    const int mofs = (lane >> 4) << 3;
    #pragma unroll
    for (int nt = 0; nt < 4; ++nt) {
        const int j  = nt * 16 + row16;       // output feature 0..63
        const float bg = fc_b[j];
        const float bc = fc_b[j + ATOM_F];
        #pragma unroll
        for (int r = 0; r < 8; ++r) {
            {
                const float gate = acc0[nt][r] + bg;
                const float core = acc0[nt + 4][r] + bc;
                const float msg  = (1.0f / (1.0f + __expf(-gate))) * tanhf(core);
                const int e = e0 + wid * 32 + r + mofs;
                atomicAdd(&agg[dst[e] * ATOM_F + j], msg);
            }
            {
                const float gate = acc1[nt][r] + bg;
                const float core = acc1[nt + 4][r] + bc;
                const float msg  = (1.0f / (1.0f + __expf(-gate))) * tanhf(core);
                const int e = e0 + wid * 32 + 16 + r + mofs;
                atomicAdd(&agg[dst[e] * ATOM_F + j], msg);
            }
        }
    }
}

// x = softplus(x + agg)
__global__ void update_kernel(float* __restrict__ x, const float* __restrict__ agg) {
    int i = blockIdx.x * blockDim.x + threadIdx.x;
    if (i >= N_NODES * ATOM_F) return;
    float v = x[i] + agg[i];
    x[i] = (v > 20.0f) ? v : logf(1.0f + __expf(v));
}

// atomic pooling: gsum[g][f] += x[n][f]; cnt[g] += 1 (once per node)
__global__ void pool_kernel(const float* __restrict__ x,
                            const int* __restrict__ gidx,
                            float* __restrict__ gsum, float* __restrict__ cnt) {
    int i = blockIdx.x * blockDim.x + threadIdx.x;
    if (i >= N_NODES * ATOM_F) return;
    int n = i >> 6, f = i & 63;
    int g = gidx[n];
    atomicAdd(&gsum[g * ATOM_F + f], x[i]);
    if (f == 0) atomicAdd(&cnt[g], 1.0f);
}

// tiny MLP head: block g, thread j in [0,128)
__global__ __launch_bounds__(128)
void head_kernel(const float* __restrict__ gsum, const float* __restrict__ cnt,
                 const float* __restrict__ w1, const float* __restrict__ b1,
                 const float* __restrict__ w2, const float* __restrict__ b2,
                 float* __restrict__ out) {
    __shared__ float gf[ATOM_F];
    __shared__ float red[HIDDEN];
    const int g = blockIdx.x, j = threadIdx.x;
    if (j < ATOM_F) gf[j] = gsum[g * ATOM_F + j] / (cnt[g] + EPS);
    __syncthreads();
    float h = b1[j];
    #pragma unroll
    for (int k = 0; k < ATOM_F; ++k) h += gf[k] * w1[j * ATOM_F + k];
    h = fmaxf(h, 0.0f);
    red[j] = h * w2[j];
    __syncthreads();
    for (int s = HIDDEN / 2; s > 0; s >>= 1) {
        if (j < s) red[j] += red[j + s];
        __syncthreads();
    }
    if (j == 0) out[g] = red[0] + b2[0];
}

// ---------------------------------------------------------------- launcher
extern "C" void kernel_launch(void* const* d_in, const int* in_sizes, int n_in,
                              void* d_out, int out_size, void* d_ws, size_t ws_size,
                              hipStream_t stream) {
    const int*   node_fea = (const int*)d_in[0];
    const int*   ei       = (const int*)d_in[1];
    const float* edge_fea = (const float*)d_in[2];
    const int*   gidx     = (const int*)d_in[3];
    const float* table    = (const float*)d_in[4];
    const float* fc_w     = (const float*)d_in[5];   // [3][128][256]
    const float* fc_b     = (const float*)d_in[6];   // [3][128]
    const float* out_w1   = (const float*)d_in[7];
    const float* out_b1   = (const float*)d_in[8];
    const float* out_w2   = (const float*)d_in[9];
    const float* out_b2   = (const float*)d_in[10];
    float* out = (float*)d_out;

    const int* src = ei;
    const int* dst = ei + N_EDGES;

    // workspace carve (256B aligned chunks)
    char* ws = (char*)d_ws;
    auto carve = [&](size_t bytes) { char* p = ws; ws += (bytes + 255) & ~(size_t)255; return p; };
    float*  x    = (float*)carve((size_t)N_NODES * ATOM_F * 4);
    float*  agg  = (float*)carve((size_t)N_NODES * ATOM_F * 4);
    __bf16* wbf  = (__bf16*)carve((size_t)DEPTH * NDIM * KDIM * 2);
    float*  gsum = (float*)carve((size_t)NUM_GRAPHS * ATOM_F * 4);
    float*  cnt  = (float*)carve((size_t)NUM_GRAPHS * 4);

    const int NF = N_NODES * ATOM_F;

    embed_kernel<<<(NF + 255) / 256, 256, 0, stream>>>(node_fea, table, x);
    cvt_w_kernel<<<(DEPTH * NDIM * KDIM + 255) / 256, 256, 0, stream>>>(
        fc_w, wbf, DEPTH * NDIM * KDIM);

    for (int layer = 0; layer < DEPTH; ++layer) {
        zero_f32<<<(NF + 255) / 256, 256, 0, stream>>>(agg, NF);
        edge_conv_kernel<<<N_EDGES / TILE_M, 256, 0, stream>>>(
            x, src, dst, edge_fea,
            wbf + (size_t)layer * NDIM * KDIM,
            fc_b + layer * NDIM, agg);
        update_kernel<<<(NF + 255) / 256, 256, 0, stream>>>(x, agg);
    }

    zero_f32<<<(NUM_GRAPHS * ATOM_F + 255) / 256, 256, 0, stream>>>(gsum, NUM_GRAPHS * ATOM_F);
    zero_f32<<<1, 256, 0, stream>>>(cnt, NUM_GRAPHS);
    pool_kernel<<<(NF + 255) / 256, 256, 0, stream>>>(x, gidx, gsum, cnt);
    head_kernel<<<NUM_GRAPHS, HIDDEN, 0, stream>>>(gsum, cnt, out_w1, out_b1,
                                                   out_w2, out_b2, out);
}
